// PsiQRHTransformerLeech_65352222376465
// MI455X (gfx1250) — compile-verified, matrix operand order checked by reading, not run on hardware
//
#include <hip/hip_runtime.h>
#include <math.h>

// ---------------------------------------------------------------------------
// Types for CDNA5 WMMA (wave32).  v_wmma_f32_16x16x32_bf16 is codegen-confirmed.
// ---------------------------------------------------------------------------
typedef __attribute__((ext_vector_type(16))) __bf16    v16bf;
typedef __attribute__((ext_vector_type(8)))  float     v8f;
typedef __attribute__((ext_vector_type(4)))  unsigned  v4u;
typedef __attribute__((ext_vector_type(4)))  float     v4f;

#define E     1024          // embedding / FFT length
#define LATN  24            // latent width

// ws layout (byte offsets, all multiples of 256 for alignment)
#define OFF_MINMAX 0        // 2 x u32 monotonic float keys
#define OFF_FR     256      // float[1024] filter real
#define OFF_FI     4608     // float[1024] filter imag
#define OFF_TW     8960     // float[512][2] forward twiddles exp(-2*pi*i*k/1024)
#define OFF_WELT   13312    // bf16 [32][1024]  = W_el^T, rows 24..31 zero
#define OFF_WLETH  79104    // bf16 [1024][32]  = W_le^T hi part, k 24..31 zero
#define OFF_WLETL  144896   // bf16 [1024][32]  = W_le^T lo residual

// ---------------------------------------------------------------------------
// helpers
// ---------------------------------------------------------------------------
__device__ inline unsigned fkey(float f) {          // order-preserving uint key
  unsigned u = __float_as_uint(f);
  return (u & 0x80000000u) ? ~u : (u | 0x80000000u);
}
__device__ inline float fdec(unsigned k) {
  return (k & 0x80000000u) ? __uint_as_float(k ^ 0x80000000u)
                           : __uint_as_float(~k);
}
// f32 pair -> packed bf16x2 in ONE VALU op.  The toolchain has no pairwise
// builtin (has_builtin==0) and scalar casts lower to half-used converts plus
// v_mov_b16 repacks, so issue the packed convert directly (mnemonic verified
// in this toolchain's own codegen).
__device__ inline unsigned pk2(float lo, float hi) {
  unsigned r;
  asm("v_cvt_pk_bf16_f32 %0, %1, %2" : "=v"(r) : "v"(lo), "v"(hi));
  return r;
}
__device__ inline unsigned short bf16s(float f) {
  union { __bf16 h; unsigned short s; } r;
  r.h = (__bf16)f;
  return r.s;
}

// Golay H masks: syn_j = OR_k(bits_k & H[j][k]),  H = [A^T | I12].
// HMASK[j] = (column j of A as 12-bit mask) | (1 << (12+j))
__device__ __constant__ unsigned HMASK[12] = {
  0x5A7u | (1u << 12), 0xAABu | (1u << 13), 0x92Du | (1u << 14),
  0x62Eu | (1u << 15), 0x5D7u | (1u << 16), 0xAD3u | (1u << 17),
  0x951u | (1u << 18), 0x658u | (1u << 19), 0x59Cu | (1u << 20),
  0xA96u | (1u << 21), 0x91Bu | (1u << 22), 0x61Du | (1u << 23)};

// ---------------------------------------------------------------------------
// Kernel 0: init min/max keys
// ---------------------------------------------------------------------------
__global__ void k_init(unsigned* mm) {
  mm[0] = fkey(INFINITY);    // running min key
  mm[1] = fkey(-INFINITY);   // running max key
}

// ---------------------------------------------------------------------------
// Kernel 1: global min/max of x (16.7M f32) — pure bandwidth, one pass
// ---------------------------------------------------------------------------
__global__ __launch_bounds__(256) void k_minmax(const float* __restrict__ x,
                                                unsigned n, unsigned* mm) {
  __shared__ unsigned smin[256], smax[256];
  unsigned t = threadIdx.x;
  unsigned kmin = 0xFFFFFFFFu, kmax = 0u;
  for (unsigned i = blockIdx.x * 256u + t; i < n; i += gridDim.x * 256u) {
    unsigned k = fkey(x[i]);
    kmin = k < kmin ? k : kmin;
    kmax = k > kmax ? k : kmax;
  }
  smin[t] = kmin; smax[t] = kmax;
  __syncthreads();
  for (unsigned s = 128; s > 0; s >>= 1) {
    if (t < s) {
      smin[t] = smin[t + s] < smin[t] ? smin[t + s] : smin[t];
      smax[t] = smax[t + s] > smax[t] ? smax[t + s] : smax[t];
    }
    __syncthreads();
  }
  if (t == 0) { atomicMin(&mm[0], smin[0]); atomicMax(&mm[1], smax[0]); }
}

// ---------------------------------------------------------------------------
// Kernel 2: fractal dimension D, wave filter, twiddles, bf16 weight transposes
// ---------------------------------------------------------------------------
__global__ __launch_bounds__(256) void k_prep(const float* __restrict__ x,
                                              const float* __restrict__ W_el,
                                              const float* __restrict__ W_le,
                                              unsigned char* ws) {
  __shared__ float sD;
  unsigned t = threadIdx.x;
  if (t == 0) {
    const unsigned* mm = (const unsigned*)(ws + OFF_MINMAX);
    float mn = fdec(mm[0]), mx = fdec(mm[1]);
    float inv = 1.0f / (mx - mn + 1e-10f);
    float sx = 0.f, sy = 0.f, sxy = 0.f, sxx = 0.f;
    const int scs[5] = {2, 4, 8, 16, 32};
    for (int s = 0; s < 5; ++s) {
      int cnt = scs[s] * scs[s];
      float sum = 0.f;
      for (int j = 0; j < cnt; ++j) sum += (x[j] - mn) * inv;
      float mean = sum / (float)cnt;
      float c = 0.f;
      for (int j = 0; j < cnt; ++j)
        if ((x[j] - mn) * inv > mean) c += 1.0f;
      if (c < 1.0f) c = 1.0f;
      float lx = logf((float)scs[s]), ly = logf(c);
      sx += lx; sy += ly; sxy += lx * ly; sxx += lx * lx;
    }
    float num = 5.f * sxy - sx * sy;
    float den = 5.f * sxx - sx * sx;
    sD = fminf(fmaxf(-num / den, 1.0f), 2.0f);
  }
  __syncthreads();
  float D = sD;
  float omega = 1.0f + 0.2f * D, aa = 1.0f + 0.5f * D, bb = 0.1f * D;
  float* fr = (float*)(ws + OFF_FR);
  float* fi = (float*)(ws + OFF_FI);
  for (int k = t; k < E; k += 256) {
    float f  = (k < E / 2 ? (float)k : (float)(k - E)) * (1.0f / (float)E);
    float ka = fabsf(f) + 1e-8f;
    float re = sinf(omega * D + aa * ka);
    float ph = omega * D - ka + bb * ka * ka;
    fr[k] = re * cosf(ph);
    fi[k] = re * sinf(ph);
  }
  float* tw = (float*)(ws + OFF_TW);
  const float w0 = -6.283185307179586f / (float)E;
  for (int k = t; k < E / 2; k += 256) {
    tw[2 * k]     = cosf(w0 * (float)k);
    tw[2 * k + 1] = sinf(w0 * (float)k);
  }
  // W_el (E x 24) -> bf16 [n][k], n padded to 32 with zeros
  unsigned short* welt = (unsigned short*)(ws + OFF_WELT);
  for (int i = t; i < 32 * E; i += 256) {
    int n = i >> 10, k = i & (E - 1);
    float v = (n < LATN) ? W_el[k * LATN + n] : 0.0f;
    welt[i] = bf16s(v);
  }
  // W_le (24 x E) -> bf16 [n][k] hi + residual lo, k padded to 32
  unsigned short* wh = (unsigned short*)(ws + OFF_WLETH);
  unsigned short* wl = (unsigned short*)(ws + OFF_WLETL);
  for (int i = t; i < E * 32; i += 256) {
    int n = i >> 5, k = i & 31;
    float v = (k < LATN) ? W_le[k * E + n] : 0.0f;
    __bf16 hb = (__bf16)v;
    union { __bf16 h; unsigned short s; } hu; hu.h = hb;
    wh[i] = hu.s;
    wl[i] = bf16s(v - (float)hb);
  }
}

// ---------------------------------------------------------------------------
// Kernel 3: per-row  x1 = Re(IFFT(FFT(x) * filt)).   One 256-thread block per
// row; 1024-pt radix-2 in LDS.  Forward = DIT (bit-rev in), inverse = DIF
// (bit-rev out) so no mid-pipeline reorder copy is needed.
// The row is staged into LDS with gfx1250 async global->LDS DMA, each lane
// scattering its element straight into the bit-reversed slot (per-lane LDS
// destination = low 32 bits of the generic pointer, per ISA aperture rule).
// ---------------------------------------------------------------------------
__global__ __launch_bounds__(256) void k_fft(const float* __restrict__ x,
                                             float* __restrict__ x1,
                                             const unsigned char* __restrict__ ws) {
  __shared__ float re[E], im[E];
  __shared__ float twc[E / 2], tws[E / 2];
  const unsigned t = threadIdx.x;
  const float* twg = (const float*)(ws + OFF_TW);
  for (int k = t; k < E / 2; k += 256) { twc[k] = twg[2 * k]; tws[k] = twg[2 * k + 1]; }
  const float* row = x + (size_t)blockIdx.x * E;
#pragma unroll
  for (int j = 0; j < 4; ++j) {
    int i = t + 256 * j;
    int r = __brev((unsigned)i) >> 22;   // 10-bit reversal
    unsigned lds_dst = (unsigned)(size_t)&re[r];   // LDS byte address (addr[31:0])
    const float* gsrc = row + i;
    asm volatile("global_load_async_to_lds_b32 %0, %1, off"
                 :: "v"(lds_dst), "v"(gsrc) : "memory");
    im[r] = 0.0f;
  }
  asm volatile("s_wait_asynccnt 0" ::: "memory");  // ASYNCcnt drain (TDM-class path)
  __syncthreads();
  // forward DIT, sign -1
  for (int s = 1; s <= 10; ++s) {
    int hf = 1 << (s - 1);
#pragma unroll
    for (int it = 0; it < 2; ++it) {
      int b   = t + (it << 8);
      int pos = b & (hf - 1);
      int i0  = ((b >> (s - 1)) << s) + pos;
      int i1  = i0 + hf;
      int ti  = pos << (10 - s);
      float wr = twc[ti], wi = tws[ti];
      float ur = re[i0], ui = im[i0];
      float vr = re[i1], vi = im[i1];
      float tr = vr * wr - vi * wi;
      float tq = vr * wi + vi * wr;
      re[i0] = ur + tr; im[i0] = ui + tq;
      re[i1] = ur - tr; im[i1] = ui - tq;
    }
    __syncthreads();
  }
  // filter multiply (natural order)
  const float* fr = (const float*)(ws + OFF_FR);
  const float* fi = (const float*)(ws + OFF_FI);
#pragma unroll
  for (int j = 0; j < 4; ++j) {
    int i = t + 256 * j;
    float zr = re[i], zi = im[i];
    float cr = fr[i], ci = fi[i];
    re[i] = zr * cr - zi * ci;
    im[i] = zr * ci + zi * cr;
  }
  __syncthreads();
  // inverse DIF, sign +1 (conjugate twiddles), output bit-reversed
  for (int s = 10; s >= 1; --s) {
    int hf = 1 << (s - 1);
#pragma unroll
    for (int it = 0; it < 2; ++it) {
      int b   = t + (it << 8);
      int pos = b & (hf - 1);
      int i0  = ((b >> (s - 1)) << s) + pos;
      int i1  = i0 + hf;
      int ti  = pos << (10 - s);
      float wr = twc[ti], wi = -tws[ti];
      float ur = re[i0], ui = im[i0];
      float vr = re[i1], vi = im[i1];
      re[i0] = ur + vr; im[i0] = ui + vi;
      float dr = ur - vr, di = ui - vi;
      re[i1] = dr * wr - di * wi;
      im[i1] = dr * wi + di * wr;
    }
    __syncthreads();
  }
  float* orow = x1 + (size_t)blockIdx.x * E;
  const float scale = 1.0f / (float)E;
#pragma unroll
  for (int j = 0; j < 4; ++j) {
    int i = t + 256 * j;
    orow[i] = re[__brev((unsigned)i) >> 22] * scale;
  }
}

// ---------------------------------------------------------------------------
// Kernel 4: per 16-row block (one wave32):
//   lat = bf16(x1) @ W_el        (2 WMMA per K-step, 32 K-steps)
//   bits/Golay via bit logic in LDS
//   out = x1 + corrected @ (W_le_hi + W_le_lo) + b_le   (2 WMMA per N-tile)
// A layout per ISA: lanes 0-15 row M=lane, k = {kb..kb+7, kb+16..kb+23};
// lanes 16-31 row M=lane-16, k = {kb+8..kb+15, kb+24..kb+31}.  B assumed
// symmetric (column = lane&15, same k grouping) with K-major staging in ws.
// C/D: VGPR v -> M = v + 8*(lane>>4), N = lane&15.
// ---------------------------------------------------------------------------
union BFV { v16bf bf; v4u q[2]; unsigned u[8]; };

__global__ __launch_bounds__(32) void k_gemm_golay(
    float* __restrict__ out,             // holds x1, rewritten in place
    const unsigned char* __restrict__ ws,
    const float* __restrict__ b_el, const float* __restrict__ b_le) {
  __shared__ float    lat[16 * 32];
  __shared__ unsigned msk[16];
  const int lane = threadIdx.x;
  const int hf   = lane >> 4;
  const int l16  = lane & 15;
  const int row0 = blockIdx.x * 16;

  const unsigned short* welt = (const unsigned short*)(ws + OFF_WELT);
  const float* xr = out + (size_t)(row0 + l16) * E;

  v8f c0 = {0.f, 0.f, 0.f, 0.f, 0.f, 0.f, 0.f, 0.f};
  v8f c1 = c0;
  for (int kt = 0; kt < 32; ++kt) {
    int kb = kt * 32 + hf * 8;
    if (kt + 1 < 32) __builtin_prefetch(xr + kb + 32, 0, 3);  // near-cache prefetch
    v4f a0 = ((const v4f*)(xr + kb))[0];
    v4f a1 = ((const v4f*)(xr + kb))[1];
    v4f a2 = ((const v4f*)(xr + kb + 16))[0];
    v4f a3 = ((const v4f*)(xr + kb + 16))[1];
    BFV A;
    A.u[0] = pk2(a0.x, a0.y); A.u[1] = pk2(a0.z, a0.w);
    A.u[2] = pk2(a1.x, a1.y); A.u[3] = pk2(a1.z, a1.w);
    A.u[4] = pk2(a2.x, a2.y); A.u[5] = pk2(a2.z, a2.w);
    A.u[6] = pk2(a3.x, a3.y); A.u[7] = pk2(a3.z, a3.w);
    BFV B0, B1;
    B0.q[0] = ((const v4u*)(welt + (size_t)l16 * E + kb))[0];
    B0.q[1] = ((const v4u*)(welt + (size_t)l16 * E + kb + 16))[0];
    B1.q[0] = ((const v4u*)(welt + (size_t)(16 + l16) * E + kb))[0];
    B1.q[1] = ((const v4u*)(welt + (size_t)(16 + l16) * E + kb + 16))[0];
    c0 = __builtin_amdgcn_wmma_f32_16x16x32_bf16(false, A.bf, false, B0.bf,
                                                 (short)0, c0, false, false);
    c1 = __builtin_amdgcn_wmma_f32_16x16x32_bf16(false, A.bf, false, B1.bf,
                                                 (short)0, c1, false, false);
  }
#pragma unroll
  for (int v = 0; v < 8; ++v) {
    int m = v + 8 * hf;
    lat[m * 32 + l16]      = c0[v];
    lat[m * 32 + 16 + l16] = c1[v];
  }
  __syncthreads();
  if (lane < 16) {                       // one row per lane: bits + Golay
    unsigned mask = 0;
#pragma unroll
    for (int k = 0; k < 24; ++k)
      if (lat[lane * 32 + k] + b_el[k] > 0.0f) mask |= (1u << k);
    unsigned syn = 0;
#pragma unroll
    for (int j = 0; j < 12; ++j)
      if (mask & HMASK[j]) syn |= (1u << j);
    int w = __popc(syn);
    if (w >= 1 && w <= 3) mask ^= syn;   // flip syndrome-set positions (<12)
    msk[lane] = mask;
  }
  __syncthreads();
  BFV A2;                                // corrected bits as exact bf16 0/1
  {
    unsigned m = msk[l16];
#pragma unroll
    for (int p = 0; p < 8; ++p) {
      int k0 = (p < 4) ? (hf * 8 + 2 * p) : (16 + hf * 8 + 2 * (p - 4));
      unsigned lo = ((k0 < 24)     && ((m >> k0) & 1u))       ? 0x3F80u : 0u;
      unsigned hi = ((k0 + 1 < 24) && ((m >> (k0 + 1)) & 1u)) ? 0x3F80u : 0u;
      A2.u[p] = lo | (hi << 16);
    }
  }
  const unsigned short* wh = (const unsigned short*)(ws + OFF_WLETH);
  const unsigned short* wl = (const unsigned short*)(ws + OFF_WLETL);
  for (int nt = 0; nt < 64; ++nt) {
    int n = nt * 16 + l16;
    BFV Bh, Bl;
    Bh.q[0] = ((const v4u*)(wh + n * 32 + hf * 8))[0];
    Bh.q[1] = ((const v4u*)(wh + n * 32 + hf * 8 + 16))[0];
    Bl.q[0] = ((const v4u*)(wl + n * 32 + hf * 8))[0];
    Bl.q[1] = ((const v4u*)(wl + n * 32 + hf * 8 + 16))[0];
    v8f c = {0.f, 0.f, 0.f, 0.f, 0.f, 0.f, 0.f, 0.f};
    c = __builtin_amdgcn_wmma_f32_16x16x32_bf16(false, A2.bf, false, Bh.bf,
                                                (short)0, c, false, false);
    c = __builtin_amdgcn_wmma_f32_16x16x32_bf16(false, A2.bf, false, Bl.bf,
                                                (short)0, c, false, false);
    float bl = b_le[n];
#pragma unroll
    for (int v = 0; v < 8; ++v) {
      size_t idx = (size_t)(row0 + v + 8 * hf) * E + n;
      out[idx] = out[idx] + c[v] + bl;   // x1 + correction + bias, in place
    }
  }
}

// ---------------------------------------------------------------------------
extern "C" void kernel_launch(void* const* d_in, const int* in_sizes, int n_in,
                              void* d_out, int out_size, void* d_ws, size_t ws_size,
                              hipStream_t stream) {
  const float* x    = (const float*)d_in[0];
  const float* W_el = (const float*)d_in[1];
  const float* b_el = (const float*)d_in[2];
  const float* W_le = (const float*)d_in[3];
  const float* b_le = (const float*)d_in[4];
  float* out        = (float*)d_out;
  unsigned char* ws = (unsigned char*)d_ws;

  const unsigned n   = (unsigned)in_sizes[0];   // B*S*E
  const int     rows = (int)(n / E);            // B*S

  k_init  <<<1, 1, 0, stream>>>((unsigned*)(ws + OFF_MINMAX));
  k_minmax<<<1024, 256, 0, stream>>>(x, n, (unsigned*)(ws + OFF_MINMAX));
  k_prep  <<<1, 256, 0, stream>>>(x, W_el, W_le, ws);
  k_fft   <<<rows, 256, 0, stream>>>(x, out, ws);
  k_gemm_golay<<<rows / 16, 32, 0, stream>>>(out, ws, b_el, b_le);
}
